// GRACE_66941360276190
// MI455X (gfx1250) — compile-verified, matrix operand order checked by reading, not run on hardware
//
#include <hip/hip_runtime.h>
#include <hip/hip_bf16.h>
#include <cstdint>
#include <cstddef>

typedef __attribute__((ext_vector_type(16))) __bf16 v16bf;
typedef __attribute__((ext_vector_type(8)))  float  v8f;
typedef __attribute__((ext_vector_type(4)))  int    v4i;

#define NROWS 8192
#define DIM   512
#define NTOT  (2 * NROWS)        // z1 and z2 rows processed together
#define TAU_INV 2.0f             // 1 / 0.5
#define BM 128
#define BN 128
#define BK 32
#define LDST 40                  // LDS row stride (elements): 80B, 16B aligned, bank-skewed
#define THREADS 256
#define TILE_ELEMS (BM * LDST)   // one A or B tile in LDS

struct Raw32 { uint4 a, b; };

// ---------------------------------------------------------------------------
// CDNA5 async global->LDS copy (ASYNCcnt path), 16 bytes per lane.
// Builtin signature (from probe diagnostics): (v4i AS1*, v4i AS3*, Ii, Ii)
// ---------------------------------------------------------------------------
__device__ __forceinline__ void async_copy16(const __bf16* g, __bf16* l) {
#if __has_builtin(__builtin_amdgcn_global_load_async_to_lds_b128)
  void* gp = const_cast<void*>((const void*)g);
  auto g1 = (__attribute__((address_space(1))) v4i*)gp;
  auto l3 = (__attribute__((address_space(3))) v4i*)(void*)l;
  __builtin_amdgcn_global_load_async_to_lds_b128(g1, l3, 0, 0);
#else
  asm volatile("global_load_async_to_lds_b128 %0, %1, off"
               :: "v"((unsigned)(size_t)(__attribute__((address_space(3))) __bf16*)l),
                  "v"(g)
               : "memory");
#endif
}

template <int N>
__device__ __forceinline__ void wait_asynccnt() {
#if __has_builtin(__builtin_amdgcn_s_wait_asynccnt)
  __builtin_amdgcn_s_wait_asynccnt((unsigned short)N);
#else
  if constexpr (N == 0) asm volatile("s_wait_asynccnt 0x0" ::: "memory");
  else                  asm volatile("s_wait_asynccnt 0x2" ::: "memory");
#endif
}

// ---------------------------------------------------------------------------
// Fragment load: 16-bit A/B operand for v_wmma_*_16x16x32_*.
// lane l holds row/col (l & 15); K-halves split by (l & 16): kbase 0 or 8,
// elements [kbase, kbase+8) and [kbase+16, kbase+24)  -> two 16B LDS loads.
// ---------------------------------------------------------------------------
__device__ __forceinline__ v16bf load_frag(const __bf16* lds, int strip, int lane) {
  const int r  = strip + (lane & 15);
  const int kb = (lane & 16) ? 8 : 0;
  Raw32 raw;
  raw.a = *(const uint4*)(lds + r * LDST + kb);
  raw.b = *(const uint4*)(lds + r * LDST + kb + 16);
  return __builtin_bit_cast(v16bf, raw);
}

// ---------------------------------------------------------------------------
// Shared mainloop: C(128x128) += A(rowBase..+128, :512) * B(colBase..+128, :512)^T
// 8 waves; wave w owns rows [w*16, w*16+16) x all 128 cols (8 wmma tiles).
// Double-buffered LDS; tiles stream in via async global->LDS loads so the
// next tile's fetch overlaps the current tile's WMMAs.
// ---------------------------------------------------------------------------
__device__ __forceinline__ void mma_mainloop(const __bf16* __restrict__ A,
                                             const __bf16* __restrict__ B,
                                             int rowBase, int colBase,
                                             __bf16* lds,   // 4 * TILE_ELEMS
                                             v8f acc[8]) {
  const int t    = threadIdx.x;
  const int lane = t & 31;
  const int w    = t >> 5;
  const int lr   = t >> 1;          // 0..127: tile row this thread stages
  const int lh   = (t & 1) << 4;    // element offset 0 or 16 within 32-elem chunk
  const __bf16* gA = A + (size_t)(rowBase + lr) * DIM + lh;
  const __bf16* gB = B + (size_t)(colBase + lr) * DIM + lh;
  __bf16* const sA[2] = { lds,                  lds + 2 * TILE_ELEMS };
  __bf16* const sB[2] = { lds + TILE_ELEMS,     lds + 3 * TILE_ELEMS };
  const int stg = lr * LDST + lh;   // this thread's staging slot inside a tile

  // prologue: stage tile 0 into buffer 0 (ASYNCcnt += 2)
  async_copy16(gA, sA[0] + stg);
  async_copy16(gB, sB[0] + stg);

  constexpr int NK = DIM / BK;      // 16 K-steps
  for (int k = 0; k < NK; ++k) {
    const int cur = k & 1;
    if (k + 1 < NK) {
      // buffer cur^1 was last read in iteration k-1; trailing barrier protects it
      async_copy16(gA + (k + 1) * BK, sA[cur ^ 1] + stg);
      async_copy16(gB + (k + 1) * BK, sB[cur ^ 1] + stg);
      wait_asynccnt<2>();           // in-order: tile k is complete, tile k+1 in flight
    } else {
      wait_asynccnt<0>();
    }
    __syncthreads();                // everyone's tile-k data visible in LDS
    const v16bf fa = load_frag(sA[cur], w * 16, lane);
#pragma unroll
    for (int j = 0; j < 8; ++j) {
      const v16bf fb = load_frag(sB[cur], j * 16, lane);
      acc[j] = __builtin_amdgcn_wmma_f32_16x16x32_bf16(false, fa, false, fb,
                                                       (short)0, acc[j], false, false);
    }
    __syncthreads();                // all reads of buffer cur done before reuse
  }
}

// ---------------------------------------------------------------------------
// Layer 1: mid = ELU(Z @ W1^T + b1), stored bf16
// ---------------------------------------------------------------------------
__global__ void __launch_bounds__(THREADS)
gemm_elu_kernel(const __bf16* __restrict__ A, const __bf16* __restrict__ W,
                const float* __restrict__ bias, __bf16* __restrict__ out) {
  __shared__ __align__(16) __bf16 lds[4 * TILE_ELEMS];
  const int rowBase = blockIdx.x * BM;
  const int colBase = blockIdx.y * BN;
  v8f acc[8] = {};
  mma_mainloop(A, W, rowBase, colBase, lds, acc);

  const int lane = threadIdx.x & 31;
  const int w    = threadIdx.x >> 5;
  const int rw   = rowBase + w * 16 + ((lane & 16) ? 8 : 0);
  const int cl   = colBase + (lane & 15);
#pragma unroll
  for (int j = 0; j < 8; ++j) {
    const int col = cl + j * 16;
    const float b = bias[col];
#pragma unroll
    for (int r = 0; r < 8; ++r) {
      float v = acc[j][r] + b;
      v = v > 0.0f ? v : expm1f(v);
      out[(size_t)(rw + r) * DIM + col] = (__bf16)v;
    }
  }
}

// ---------------------------------------------------------------------------
// Layer 2: H = mid @ W2^T + b2 (f32), fused per-row sum of squares -> sq[]
// ---------------------------------------------------------------------------
__global__ void __launch_bounds__(THREADS)
gemm_bias_kernel(const __bf16* __restrict__ A, const __bf16* __restrict__ W,
                 const float* __restrict__ bias, float* __restrict__ H,
                 float* __restrict__ sq) {
  __shared__ __align__(16) __bf16 lds[4 * TILE_ELEMS];
  const int rowBase = blockIdx.x * BM;
  const int colBase = blockIdx.y * BN;
  v8f acc[8] = {};
  mma_mainloop(A, W, rowBase, colBase, lds, acc);

  const int lane = threadIdx.x & 31;
  const int w    = threadIdx.x >> 5;
  const int rw   = rowBase + w * 16 + ((lane & 16) ? 8 : 0);
  const int cl   = colBase + (lane & 15);
  float rs[8];
#pragma unroll
  for (int r = 0; r < 8; ++r) rs[r] = 0.0f;
#pragma unroll
  for (int j = 0; j < 8; ++j) {
    const int col = cl + j * 16;
    const float b = bias[col];
#pragma unroll
    for (int r = 0; r < 8; ++r) {
      float v = acc[j][r] + b;
      H[(size_t)(rw + r) * DIM + col] = v;
      float s = v * v;                    // partial ||row||^2 over these 16 cols
      s += __shfl_xor(s, 1, 32);
      s += __shfl_xor(s, 2, 32);
      s += __shfl_xor(s, 4, 32);
      s += __shfl_xor(s, 8, 32);
      rs[r] += s;
    }
  }
  if ((lane & 15) == 0) {
#pragma unroll
    for (int r = 0; r < 8; ++r) atomicAdd(&sq[rw + r], rs[r]);
  }
}

// ---------------------------------------------------------------------------
// Row-normalize H -> bf16 zn (F.normalize with eps=1e-12)
// ---------------------------------------------------------------------------
__global__ void normalize_kernel(const float* __restrict__ H,
                                 const float* __restrict__ sq,
                                 __bf16* __restrict__ zn) {
  const int idx = blockIdx.x * blockDim.x + threadIdx.x;
  const int row = idx >> 9;                         // DIM == 512
  const float n = fmaxf(sqrtf(sq[row]), 1e-12f);
  zn[idx] = (__bf16)(H[idx] / n);
}

// ---------------------------------------------------------------------------
// Similarity tile: never materialize S. For a 128x128 tile of exp((A B^T)/tau):
//   rowAcc[i] += row sums, colAcc[j] += col sums (transpose credit / l2 path),
//   diag[i]    = diagonal entries (bi==bj tiles).
// symmetric=1: only bi<=bj tiles run; off-diag tiles credit rows bj via colAcc.
// ---------------------------------------------------------------------------
__global__ void __launch_bounds__(THREADS)
sim_kernel(const __bf16* __restrict__ A, const __bf16* __restrict__ B,
           float* __restrict__ rowAcc, float* __restrict__ colAcc,
           float* __restrict__ diag, int symmetric) {
  const int bi = blockIdx.x, bj = blockIdx.y;
  if (symmetric && bi > bj) return;
  __shared__ __align__(16) __bf16 lds[4 * TILE_ELEMS];
  v8f acc[8] = {};
  mma_mainloop(A, B, bi * BM, bj * BN, lds, acc);

  const int lane = threadIdx.x & 31;
  const int w    = threadIdx.x >> 5;

  // exp(S / tau) in-register
#pragma unroll
  for (int j = 0; j < 8; ++j)
#pragma unroll
    for (int r = 0; r < 8; ++r)
      acc[j][r] = __expf(acc[j][r] * TAU_INV);

  // diagonal capture (tile j == w holds the diagonal sub-tile)
  if (diag != nullptr && bi == bj) {
#pragma unroll
    for (int j = 0; j < 8; ++j) if (j == w) {
      const int gr = bi * BM + w * 16;
#pragma unroll
      for (int r = 0; r < 8; ++r) {
        if (lane == r)      diag[gr + r]     = acc[j][r];   // lanes 0-15 half
        if (lane == r + 24) diag[gr + 8 + r] = acc[j][r];   // lanes 16-31 half
      }
    }
  }

  // row sums: each row's 16 cols live in one VGPR across a 16-lane half
  float rs[8];
#pragma unroll
  for (int r = 0; r < 8; ++r) rs[r] = 0.0f;
#pragma unroll
  for (int j = 0; j < 8; ++j)
#pragma unroll
    for (int r = 0; r < 8; ++r) {
      float v = acc[j][r];
      v += __shfl_xor(v, 1, 32);
      v += __shfl_xor(v, 2, 32);
      v += __shfl_xor(v, 4, 32);
      v += __shfl_xor(v, 8, 32);
      rs[r] += v;
    }
  if ((lane & 15) == 0) {
    const int rb = bi * BM + w * 16 + ((lane & 16) ? 8 : 0);
#pragma unroll
    for (int r = 0; r < 8; ++r) atomicAdd(&rowAcc[rb + r], rs[r]);
  }

  // column sums (skip on symmetric diagonal tiles to avoid double count)
  if (!symmetric || bi != bj) {
#pragma unroll
    for (int j = 0; j < 8; ++j) {
      float c = 0.0f;
#pragma unroll
      for (int r = 0; r < 8; ++r) c += acc[j][r];
      c += __shfl_xor(c, 16, 32);
      if (lane < 16) atomicAdd(&colAcc[bj * BN + j * 16 + lane], c);
    }
  }
}

// ---------------------------------------------------------------------------
// f32 -> bf16 conversion
// ---------------------------------------------------------------------------
__global__ void f32_to_bf16_kernel(const float* __restrict__ src,
                                   __bf16* __restrict__ dst, int n) {
  const int i = blockIdx.x * blockDim.x + threadIdx.x;
  if (i < n) dst[i] = (__bf16)src[i];
}

// ---------------------------------------------------------------------------
// Final loss: mean_i 0.5 * (log(den1/p) + log(den2/p))
// ---------------------------------------------------------------------------
__global__ void loss_kernel(const float* __restrict__ r11, const float* __restrict__ r22,
                            const float* __restrict__ r12r, const float* __restrict__ r12c,
                            const float* __restrict__ d11, const float* __restrict__ d22,
                            const float* __restrict__ d12, float* __restrict__ out) {
  __shared__ float red[256];
  float s = 0.0f;
  for (int i = threadIdx.x; i < NROWS; i += 256) {
    const float p    = d12[i];
    const float den1 = r11[i] + r12r[i] - d11[i];
    const float den2 = r22[i] + r12c[i] - d22[i];
    s += 0.5f * (logf(den1 / p) + logf(den2 / p));
  }
  red[threadIdx.x] = s;
  __syncthreads();
  for (int st = 128; st > 0; st >>= 1) {
    if (threadIdx.x < st) red[threadIdx.x] += red[threadIdx.x + st];
    __syncthreads();
  }
  if (threadIdx.x == 0) out[0] = red[0] / (float)NROWS;
}

// ---------------------------------------------------------------------------
// Host side
// ---------------------------------------------------------------------------
extern "C" void kernel_launch(void* const* d_in, const int* in_sizes, int n_in,
                              void* d_out, int out_size, void* d_ws, size_t ws_size,
                              hipStream_t stream) {
  const float* z1 = (const float*)d_in[0];
  const float* z2 = (const float*)d_in[1];
  const float* w1 = (const float*)d_in[2];
  const float* b1 = (const float*)d_in[3];
  const float* w2 = (const float*)d_in[4];
  const float* b2 = (const float*)d_in[5];

  char* ws = (char*)d_ws;
  constexpr size_t SZ_ZB  = (size_t)NTOT * DIM * 2;       // 16 MB bf16 inputs
  constexpr size_t SZ_W   = (size_t)DIM * DIM * 2;        // 0.5 MB per weight
  constexpr size_t SZ_MID = (size_t)NTOT * DIM * 2;       // 16 MB bf16 mid
  constexpr size_t SZ_H   = (size_t)NTOT * DIM * 4;       // 32 MB f32 h
  constexpr size_t SZ_ZN  = (size_t)NTOT * DIM * 2;       // 16 MB bf16 normalized

  __bf16* zb   = (__bf16*)(ws);
  __bf16* w1b  = (__bf16*)(ws + SZ_ZB);
  __bf16* w2b  = (__bf16*)(ws + SZ_ZB + SZ_W);
  __bf16* midb = (__bf16*)(ws + SZ_ZB + 2 * SZ_W);
  float*  h    = (float*)(ws + SZ_ZB + 2 * SZ_W + SZ_MID);
  __bf16* zn   = (__bf16*)(ws + SZ_ZB + 2 * SZ_W + SZ_MID + SZ_H);
  float*  sq   = (float*)(ws + SZ_ZB + 2 * SZ_W + SZ_MID + SZ_H + SZ_ZN);
  float*  accf = sq + NTOT;                               // 7 x NROWS accumulators
  float* r11  = accf + 0 * NROWS;
  float* r22  = accf + 1 * NROWS;
  float* r12r = accf + 2 * NROWS;
  float* r12c = accf + 3 * NROWS;
  float* d11  = accf + 4 * NROWS;
  float* d22  = accf + 5 * NROWS;
  float* d12  = accf + 6 * NROWS;

  const __bf16* zn1 = zn;
  const __bf16* zn2 = zn + (size_t)NROWS * DIM;

  // f32 -> bf16 conversions
  f32_to_bf16_kernel<<<(NROWS * DIM) / 256, 256, 0, stream>>>(z1, zb, NROWS * DIM);
  f32_to_bf16_kernel<<<(NROWS * DIM) / 256, 256, 0, stream>>>(z2, zb + (size_t)NROWS * DIM, NROWS * DIM);
  f32_to_bf16_kernel<<<(DIM * DIM) / 256, 256, 0, stream>>>(w1, w1b, DIM * DIM);
  f32_to_bf16_kernel<<<(DIM * DIM) / 256, 256, 0, stream>>>(w2, w2b, DIM * DIM);

  // zero the atomically-accumulated buffers (sq + 7 reduction vectors)
  (void)hipMemsetAsync(sq, 0, (size_t)(NTOT + 7 * NROWS) * sizeof(float), stream);

  // projection MLP (z1 and z2 batched as 16384 rows)
  gemm_elu_kernel<<<dim3(NTOT / BM, DIM / BN), THREADS, 0, stream>>>(zb, w1b, b1, midb);
  gemm_bias_kernel<<<dim3(NTOT / BM, DIM / BN), THREADS, 0, stream>>>(midb, w2b, b2, h, sq);
  normalize_kernel<<<(NTOT * DIM) / 256, 256, 0, stream>>>(h, sq, zn);

  // similarity reductions (S11, S22 symmetric-halved; S12 full)
  sim_kernel<<<dim3(NROWS / BM, NROWS / BN), THREADS, 0, stream>>>(zn1, zn1, r11, r11, d11, 1);
  sim_kernel<<<dim3(NROWS / BM, NROWS / BN), THREADS, 0, stream>>>(zn2, zn2, r22, r22, d22, 1);
  sim_kernel<<<dim3(NROWS / BM, NROWS / BN), THREADS, 0, stream>>>(zn1, zn2, r12r, r12c, d12, 0);

  loss_kernel<<<1, 256, 0, stream>>>(r11, r22, r12r, r12c, d11, d22, d12, (float*)d_out);
}
